// CrystalFormer_lays_45045617000605
// MI455X (gfx1250) — compile-verified, matrix operand order checked by reading, not run on hardware
//
#include <hip/hip_runtime.h>
#include <cstdint>

// ---------------------------------------------------------------------------
// CDNA5 (gfx1250) transformer encoder forward.
// - All GEMMs (QKV/O projections, FFN, QK^T, P*V) on v_wmma_f32_16x16x32_bf16.
// - Tile movement via GLOBAL_LOAD_ASYNC_TO_LDS_B128 (ASYNCcnt) with
//   double-buffered LDS in both the GEMM and flash-attention kernels.
// wave32; block = 256 threads = 8 waves.
// ---------------------------------------------------------------------------

typedef unsigned short u16;
typedef __attribute__((ext_vector_type(16))) __bf16 v16bf;
typedef __attribute__((ext_vector_type(8)))  float  v8f;

__device__ __forceinline__ u16 f2bf(float f) {
    unsigned int u = __float_as_uint(f);
    u = (u + 0x7FFFu + ((u >> 16) & 1u)) >> 16;
    return (u16)u;
}

__device__ __forceinline__ v8f vzero() {
    v8f z = {0.f, 0.f, 0.f, 0.f, 0.f, 0.f, 0.f, 0.f};
    return z;
}

// Low 32 bits of a generic pointer to __shared__ = LDS byte offset
// (addrspace(3) offset occupies the low word of the flat address).
__device__ __forceinline__ unsigned lds_off_u32(const void* p) {
    return (unsigned)(unsigned long long)p;
}

// GLOBAL_LOAD_ASYNC_TO_LDS_B128: per-lane 16B global -> LDS, tracked by ASYNCcnt.
__device__ __forceinline__ void async_load_b128(unsigned lds_off, const void* gaddr) {
    asm volatile("global_load_async_to_lds_b128 %0, %1, off"
                 :: "v"(lds_off), "v"(gaddr)
                 : "memory");
}

template <int N>
__device__ __forceinline__ void wait_asynccnt() {
#if __has_builtin(__builtin_amdgcn_s_wait_asynccnt)
    __builtin_amdgcn_s_wait_asynccnt(N);
#else
    asm volatile("s_wait_asynccnt %0" :: "i"(N) : "memory");
#endif
}

__device__ __forceinline__ void wait_dscnt0() {
#if __has_builtin(__builtin_amdgcn_s_wait_dscnt)
    __builtin_amdgcn_s_wait_dscnt(0);
#else
    asm volatile("s_wait_dscnt 0" ::: "memory");
#endif
}

// A/B fragment loader for V_WMMA_F32_16X16X32_BF16.
// A-matrix 16x32 (MxK) per ISA: lane(0-15) row=lane, K = {0..7, 16..23};
// lane(16-31) row=lane-16, K = {8..15, 24..31}.  Two b128 loads per lane.
// B operand is staged transposed ([N][K]) so the same loader serves both.
// base must be 16B aligned, stride_el a multiple of 8.
__device__ __forceinline__ v16bf load_frag(const u16* base, int stride_el) {
    const int lane = threadIdx.x & 31;
    const int lo = lane & 15, hi = lane >> 4;
    const u16* p = base + lo * stride_el + hi * 8;
    union { uint4 q[2]; v16bf v; } u;
    u.q[0] = *reinterpret_cast<const uint4*>(p);
    u.q[1] = *reinterpret_cast<const uint4*>(p + 16);
    return u.v;
}

__device__ __forceinline__ v8f wmma_bf16(v16bf a, v16bf b, v8f c) {
    return __builtin_amdgcn_wmma_f32_16x16x32_bf16(
        false, a, false, b, (short)0, c, false, false);
}

__device__ __forceinline__ float gelu_exact(float x) {
    return 0.5f * x * (1.f + erff(x * 0.70710678118654752f));
}

// ---------------------------------------------------------------------------
// LayerNorm: one row (D=1024) per 256-thread block, bf16 output.
// ---------------------------------------------------------------------------
__global__ __launch_bounds__(256) void layernorm_kernel(
    const float* __restrict__ X, const float* __restrict__ g,
    const float* __restrict__ bta, u16* __restrict__ out) {
    const int row = blockIdx.x;
    const float4 x = reinterpret_cast<const float4*>(X + (size_t)row * 1024)[threadIdx.x];
    float s = x.x + x.y + x.z + x.w;
    float q = x.x * x.x + x.y * x.y + x.z * x.z + x.w * x.w;
    for (int off = 16; off >= 1; off >>= 1) {
        s += __shfl_xor(s, off);
        q += __shfl_xor(q, off);
    }
    __shared__ float rs[8], rq[8];
    if ((threadIdx.x & 31) == 0) { rs[threadIdx.x >> 5] = s; rq[threadIdx.x >> 5] = q; }
    __syncthreads();
    float ts = 0.f, tq = 0.f;
#pragma unroll
    for (int i = 0; i < 8; ++i) { ts += rs[i]; tq += rq[i]; }
    const float mu = ts * (1.f / 1024.f);
    const float var = tq * (1.f / 1024.f) - mu * mu;
    const float inv = rsqrtf(var + 1e-5f);
    const int c = threadIdx.x * 4;
    const float xv[4] = {x.x, x.y, x.z, x.w};
#pragma unroll
    for (int j = 0; j < 4; ++j)
        out[(size_t)row * 1024 + c + j] = f2bf((xv[j] - mu) * inv * g[c + j] + bta[c + j]);
}

// ---------------------------------------------------------------------------
// Weight transpose + f32->bf16:  W[R,C] f32  ->  WT[C,R] bf16
// ---------------------------------------------------------------------------
__global__ __launch_bounds__(256) void transpose_w_kernel(
    const float* __restrict__ W, u16* __restrict__ WT, int R, int C) {
    __shared__ float t[32][33];
    const int bx = blockIdx.x * 32;  // C
    const int by = blockIdx.y * 32;  // R
    const int x = threadIdx.x & 31;
    const int y4 = threadIdx.x >> 5;
#pragma unroll
    for (int i = 0; i < 4; ++i) {
        int y = y4 + i * 8;
        t[y][x] = W[(size_t)(by + y) * C + bx + x];
    }
    __syncthreads();
#pragma unroll
    for (int i = 0; i < 4; ++i) {
        int y = y4 + i * 8;
        WT[(size_t)(bx + y) * R + by + x] = f2bf(t[x][y]);
    }
}

// V [B*S, H*DV] bf16  ->  VT [B*H, DV, S] bf16  (per-head transpose)
__global__ __launch_bounds__(256) void transpose_v_kernel(
    const u16* __restrict__ V, u16* __restrict__ VT) {
    __shared__ u16 t[32][33];
    const int bh = blockIdx.z, b = bh >> 4, h = bh & 15;
    const int s0 = blockIdx.x * 32, dv0 = blockIdx.y * 32;
    const int x = threadIdx.x & 31, y4 = threadIdx.x >> 5;
#pragma unroll
    for (int i = 0; i < 4; ++i) {
        int y = y4 + i * 8;
        t[y][x] = V[(size_t)(b * 1024 + s0 + y) * 1024 + h * 64 + dv0 + x];
    }
    __syncthreads();
#pragma unroll
    for (int i = 0; i < 4; ++i) {
        int y = y4 + i * 8;
        VT[(size_t)(bh * 64 + dv0 + y) * 1024 + s0 + x] = t[x][y];
    }
}

// ---------------------------------------------------------------------------
// GEMM: C[M,N] = A[M,K](bf16) x BT[N,K](bf16)^T + bias, optional GELU,
// optional f32 residual add, f32 and/or bf16 outputs.
// 128x128x32 tiles; 8 waves, each 64x32 via 4x2 WMMA accumulators.
// Double-buffered LDS fed by async global->LDS b128 copies.
// ---------------------------------------------------------------------------
#define LDS_STRIDE 48  // 32 bf16 of K + pad; 96B rows keep every b128 aligned

__global__ __launch_bounds__(256) void gemm_bf16_kernel(
    const u16* __restrict__ A, const u16* __restrict__ BT,
    const float* __restrict__ bias, const float* __restrict__ resid,
    float* __restrict__ outF, u16* __restrict__ outB,
    int M, int N, int K, int gelu_flag) {
    __shared__ u16 sA[2][128 * LDS_STRIDE];
    __shared__ u16 sB[2][128 * LDS_STRIDE];
    const int tid = threadIdx.x;
    const int w = tid >> 5;
    const int wm = w >> 2;  // 0..1 : 64 rows each
    const int wn = w & 3;   // 0..3 : 32 cols each
    const int m0 = blockIdx.y * 128;
    const int n0 = blockIdx.x * 128;

    v8f acc[4][2];
#pragma unroll
    for (int mi = 0; mi < 4; ++mi) { acc[mi][0] = vzero(); acc[mi][1] = vzero(); }

    const int row = tid >> 1;
    const int cb = (tid & 1) * 16;
    const u16* gArow = A + (size_t)(m0 + row) * K + cb;
    const u16* gBrow = BT + (size_t)(n0 + row) * K + cb;
    const unsigned laA[2] = {lds_off_u32(&sA[0][row * LDS_STRIDE + cb]),
                             lds_off_u32(&sA[1][row * LDS_STRIDE + cb])};
    const unsigned laB[2] = {lds_off_u32(&sB[0][row * LDS_STRIDE + cb]),
                             lds_off_u32(&sB[1][row * LDS_STRIDE + cb])};

    auto issue_tile = [&](int k0, int buf) {
        async_load_b128(laA[buf], gArow + k0);
        async_load_b128(laA[buf] + 16, gArow + k0 + 8);
        async_load_b128(laB[buf], gBrow + k0);
        async_load_b128(laB[buf] + 16, gBrow + k0 + 8);
    };

    const int nk = K >> 5;
    issue_tile(0, 0);

    for (int i = 0; i < nk; ++i) {
        const int cur = i & 1;
        if (i + 1 < nk) {
            issue_tile((i + 1) << 5, cur ^ 1);
            wait_asynccnt<4>();   // 4 newest (next tile) may remain in flight
        } else {
            wait_asynccnt<0>();
        }
        __syncthreads();

        v16bf bf0 = load_frag(&sB[cur][(wn * 32 + 0) * LDS_STRIDE], LDS_STRIDE);
        v16bf bf1 = load_frag(&sB[cur][(wn * 32 + 16) * LDS_STRIDE], LDS_STRIDE);
#pragma unroll
        for (int mi = 0; mi < 4; ++mi) {
            v16bf af = load_frag(&sA[cur][(wm * 64 + mi * 16) * LDS_STRIDE], LDS_STRIDE);
            acc[mi][0] = wmma_bf16(af, bf0, acc[mi][0]);
            acc[mi][1] = wmma_bf16(af, bf1, acc[mi][1]);
        }
        __syncthreads();  // all reads of buf[cur] done before it is refilled
    }

    const int lane = tid & 31, lo = lane & 15, hi = lane >> 4;
#pragma unroll
    for (int mi = 0; mi < 4; ++mi)
#pragma unroll
        for (int ni = 0; ni < 2; ++ni)
#pragma unroll
            for (int r = 0; r < 8; ++r) {
                const int gr = m0 + wm * 64 + mi * 16 + r + 8 * hi;
                const int gc = n0 + wn * 32 + ni * 16 + lo;
                float v = acc[mi][ni][r] + bias[gc];
                if (gelu_flag) v = gelu_exact(v);
                if (resid) v += resid[(size_t)gr * N + gc];
                if (outF) outF[(size_t)gr * N + gc] = v;
                if (outB) outB[(size_t)gr * N + gc] = f2bf(v);
            }
}

// ---------------------------------------------------------------------------
// Flash attention: grid (S/128, B*H).  Each wave owns 16 q rows; online
// softmax over 32-key tiles; QK^T and P*V both on WMMA bf16.
// K/V tiles double-buffered via async global->LDS copies.
// ---------------------------------------------------------------------------
__global__ __launch_bounds__(256) void flash_attn_kernel(
    const u16* __restrict__ Q, const u16* __restrict__ Kb,
    const u16* __restrict__ VT, const int* __restrict__ mask,
    u16* __restrict__ O) {
    __shared__ u16 sK[2][32 * 72];    // 32 keys x 64 dk  (stride 72 el = 144B)
    __shared__ u16 sVT[2][64 * 40];   // 64 dv  x 32 keys (stride 40 el = 80B)
    __shared__ u16 sP[8][16 * 40];    // per-wave P tile 16 x 32 keys

    const int tid = threadIdx.x;
    const int w = tid >> 5;
    const int lane = tid & 31, lo = lane & 15, hi = lane >> 4;
    const int bh = blockIdx.y, b = bh >> 4, h = bh & 15;
    const int q0 = blockIdx.x * 128 + w * 16;

    const u16* qbase = Q + (size_t)(b * 1024 + q0) * 1024 + h * 64;
    const v16bf qf0 = load_frag(qbase, 1024);
    const v16bf qf1 = load_frag(qbase + 32, 1024);

    const u16* kbh = Kb + (size_t)(b * 1024) * 1024 + h * 64;
    const u16* vtb = VT + (size_t)(bh * 64) * 1024;

    v8f o[4];
#pragma unroll
    for (int i = 0; i < 4; ++i) o[i] = vzero();
    float mrow[8], lrow[8];
#pragma unroll
    for (int r = 0; r < 8; ++r) { mrow[r] = -1e30f; lrow[r] = 0.f; }

    const int key = tid >> 3, dkb = (tid & 7) * 8;
    const int dv = tid >> 2, kb2 = (tid & 3) * 8;
    const unsigned lK[2] = {lds_off_u32(&sK[0][key * 72 + dkb]),
                            lds_off_u32(&sK[1][key * 72 + dkb])};
    const unsigned lV[2] = {lds_off_u32(&sVT[0][dv * 40 + kb2]),
                            lds_off_u32(&sVT[1][dv * 40 + kb2])};

    auto issue_tile = [&](int kt, int buf) {
        async_load_b128(lK[buf], kbh + (size_t)(kt + key) * 1024 + dkb);
        async_load_b128(lV[buf], vtb + (size_t)dv * 1024 + kt + kb2);
    };

    issue_tile(0, 0);

    for (int it = 0; it < 32; ++it) {
        const int cur = it & 1;
        const int kt = it << 5;
        if (it + 1 < 32) {
            issue_tile((it + 1) << 5, cur ^ 1);
            wait_asynccnt<2>();
        } else {
            wait_asynccnt<0>();
        }
        __syncthreads();

        v8f s[2];
#pragma unroll
        for (int kg = 0; kg < 2; ++kg) {
            v16bf kf0 = load_frag(&sK[cur][kg * 16 * 72], 72);
            v16bf kf1 = load_frag(&sK[cur][kg * 16 * 72 + 32], 72);
            v8f sa = vzero();
            sa = wmma_bf16(qf0, kf0, sa);
            sa = wmma_bf16(qf1, kf1, sa);
            s[kg] = sa;
        }

        // scale + mask (branchless selects keep EXEC all-ones for next WMMA)
#pragma unroll
        for (int r = 0; r < 8; ++r) {
            const int qr = q0 + r + 8 * hi;
            const float v0 = s[0][r] * 0.125f;
            const float v1 = s[1][r] * 0.125f;
            s[0][r] = (mask[(size_t)qr * 1024 + kt + lo] == 0) ? -1e9f : v0;
            s[1][r] = (mask[(size_t)qr * 1024 + kt + 16 + lo] == 0) ? -1e9f : v1;
        }

        // online softmax: each lane's half owns rows r+8*hi — matches C layout
#pragma unroll
        for (int r = 0; r < 8; ++r) {
            float mx = fmaxf(s[0][r], s[1][r]);
            mx = fmaxf(mx, __shfl_xor(mx, 1));
            mx = fmaxf(mx, __shfl_xor(mx, 2));
            mx = fmaxf(mx, __shfl_xor(mx, 4));
            mx = fmaxf(mx, __shfl_xor(mx, 8));
            const float mn = fmaxf(mrow[r], mx);
            const float al = __expf(mrow[r] - mn);
            const float p0 = __expf(s[0][r] - mn);
            const float p1 = __expf(s[1][r] - mn);
            float ps = p0 + p1;
            ps += __shfl_xor(ps, 1);
            ps += __shfl_xor(ps, 2);
            ps += __shfl_xor(ps, 4);
            ps += __shfl_xor(ps, 8);
            lrow[r] = lrow[r] * al + ps;
            mrow[r] = mn;
            o[0][r] *= al; o[1][r] *= al; o[2][r] *= al; o[3][r] *= al;
            u16* pp = sP[w] + (r + 8 * hi) * 40;
            pp[lo] = f2bf(p0);
            pp[16 + lo] = f2bf(p1);
        }
        wait_dscnt0();  // per-wave P tile: LDS store -> b128 load ordering
        const v16bf pf = load_frag(sP[w], 40);
#pragma unroll
        for (int ni = 0; ni < 4; ++ni) {
            v16bf vf = load_frag(&sVT[cur][ni * 16 * 40], 40);
            o[ni] = wmma_bf16(pf, vf, o[ni]);
        }
        __syncthreads();  // reads of buf[cur] done before async refill
    }

#pragma unroll
    for (int ni = 0; ni < 4; ++ni)
#pragma unroll
        for (int r = 0; r < 8; ++r) {
            const int qr = q0 + r + 8 * hi;
            const float val = o[ni][r] / (lrow[r] + 1e-30f);
            O[(size_t)(b * 1024 + qr) * 1024 + h * 64 + ni * 16 + lo] = f2bf(val);
        }
}

// ---------------------------------------------------------------------------
// Host orchestration
// ---------------------------------------------------------------------------
extern "C" void kernel_launch(void* const* d_in, const int* in_sizes, int n_in,
                              void* d_out, int out_size, void* d_ws, size_t ws_size,
                              hipStream_t stream) {
    (void)in_sizes; (void)n_in; (void)out_size; (void)ws_size;
    const float* h_in  = (const float*)d_in[0];
    const int*   mask  = (const int*)d_in[1];
    const float* Wq    = (const float*)d_in[2];
    const float* bq    = (const float*)d_in[3];
    const float* Wk    = (const float*)d_in[4];
    const float* bk    = (const float*)d_in[5];
    const float* Wv    = (const float*)d_in[6];
    const float* bv    = (const float*)d_in[7];
    const float* Wo    = (const float*)d_in[8];
    const float* bo    = (const float*)d_in[9];
    const float* ln1g  = (const float*)d_in[10];
    const float* ln1b  = (const float*)d_in[11];
    const float* ln2g  = (const float*)d_in[12];
    const float* ln2b  = (const float*)d_in[13];
    const float* W1    = (const float*)d_in[14];
    const float* b1    = (const float*)d_in[15];
    const float* W2    = (const float*)d_in[16];
    const float* b2    = (const float*)d_in[17];

    const size_t MB = 1ull << 20;
    char* ws = (char*)d_ws;
    float* h   = (float*)(ws + 0 * MB);     // 16 MB : running hidden state f32
    u16*   ln  = (u16*)(ws + 16 * MB);      //  8 MB : LN output bf16
    u16*   qb  = (u16*)(ws + 24 * MB);      //  8 MB
    u16*   kbf = (u16*)(ws + 32 * MB);      //  8 MB
    u16*   vbf = (u16*)(ws + 40 * MB);      //  8 MB
    u16*   vt  = (u16*)(ws + 48 * MB);      //  8 MB : V transposed per head
    u16*   ao  = (u16*)(ws + 56 * MB);      //  8 MB : attention output bf16
    u16*   ffh = (u16*)(ws + 64 * MB);      // 32 MB : FFN hidden bf16
    u16*   wT  = (u16*)(ws + 96 * MB);      //  8 MB : transposed weight bf16

    hipMemcpyAsync(h, h_in, 4096ull * 1024 * sizeof(float),
                   hipMemcpyDeviceToDevice, stream);

    const dim3 blk(256);
    const dim3 gemm_n1k(8, 32);    // N=1024
    const dim3 gemm_n4k(32, 32);   // N=4096

    for (int l = 0; l < 4; ++l) {
        const size_t wDD = (size_t)l * 1024 * 1024;
        const size_t wDF = (size_t)l * 1024 * 4096;

        // ---- attention block ----
        layernorm_kernel<<<4096, blk, 0, stream>>>(h, ln1g + l * 1024, ln1b + l * 1024, ln);

        transpose_w_kernel<<<dim3(32, 32), blk, 0, stream>>>(Wq + wDD, wT, 1024, 1024);
        gemm_bf16_kernel<<<gemm_n1k, blk, 0, stream>>>(ln, wT, bq + l * 1024,
            nullptr, nullptr, qb, 4096, 1024, 1024, 0);
        transpose_w_kernel<<<dim3(32, 32), blk, 0, stream>>>(Wk + wDD, wT, 1024, 1024);
        gemm_bf16_kernel<<<gemm_n1k, blk, 0, stream>>>(ln, wT, bk + l * 1024,
            nullptr, nullptr, kbf, 4096, 1024, 1024, 0);
        transpose_w_kernel<<<dim3(32, 32), blk, 0, stream>>>(Wv + wDD, wT, 1024, 1024);
        gemm_bf16_kernel<<<gemm_n1k, blk, 0, stream>>>(ln, wT, bv + l * 1024,
            nullptr, nullptr, vbf, 4096, 1024, 1024, 0);

        transpose_v_kernel<<<dim3(32, 2, 64), blk, 0, stream>>>(vbf, vt);
        flash_attn_kernel<<<dim3(8, 64), blk, 0, stream>>>(qb, kbf, vt, mask, ao);

        transpose_w_kernel<<<dim3(32, 32), blk, 0, stream>>>(Wo + wDD, wT, 1024, 1024);
        gemm_bf16_kernel<<<gemm_n1k, blk, 0, stream>>>(ao, wT, bo + l * 1024,
            h, h, nullptr, 4096, 1024, 1024, 0);   // residual add -> h (f32)

        // ---- FFN block ----
        layernorm_kernel<<<4096, blk, 0, stream>>>(h, ln2g + l * 1024, ln2b + l * 1024, ln);

        transpose_w_kernel<<<dim3(128, 32), blk, 0, stream>>>(W1 + wDF, wT, 1024, 4096);
        gemm_bf16_kernel<<<gemm_n4k, blk, 0, stream>>>(ln, wT, b1 + l * 4096,
            nullptr, nullptr, ffh, 4096, 4096, 1024, 1);  // GELU

        transpose_w_kernel<<<dim3(32, 128), blk, 0, stream>>>(W2 + wDF, wT, 4096, 1024);
        gemm_bf16_kernel<<<gemm_n1k, blk, 0, stream>>>(ffh, wT, b2 + l * 1024,
            h, h, nullptr, 4096, 1024, 4096, 0);   // residual add -> h (f32)
    }

    hipMemcpyAsync(d_out, h, 4096ull * 1024 * sizeof(float),
                   hipMemcpyDeviceToDevice, stream);
}
